// NeuralJumpODE_4509715661064
// MI455X (gfx1250) — compile-verified, matrix operand order checked by reading, not run on hardware
//
#include <hip/hip_runtime.h>
#include <hip/hip_bf16.h>

// ---------------- problem constants ----------------
#define BATCH   512
#define NOBS    64
#define DX      32
#define HID     512
#define DY      32
#define KF      546          // HID + DX + 2
#define KFPAD   576          // KF rounded up to 32 (zero-padded)
#define MT      16           // batch rows per block (== WMMA M)
#define NTHREADS 512         // 16 wave32 waves
#define NWAVES  16

typedef __attribute__((ext_vector_type(16))) __bf16 v16bf;
typedef __attribute__((ext_vector_type(8)))  __bf16 v8bf;
typedef __attribute__((ext_vector_type(8)))  float  v8f;

// ws layout (bf16 elements): weights transposed to [N][Kpad], zero padded
#define OFF_JW1  ((size_t)0)
#define OFF_JW2  (OFF_JW1 + (size_t)HID * DX)     // jW1T [512][32]
#define OFF_FW1  (OFF_JW2 + (size_t)HID * HID)    // jW2T [512][512]
#define OFF_FW2  (OFF_FW1 + (size_t)HID * KFPAD)  // fW1T [512][576]
#define OFF_OW1  (OFF_FW2 + (size_t)HID * HID)    // fW2T [512][512]
#define OFF_OW2  (OFF_OW1 + (size_t)HID * HID)    // oW1T [512][512]
#define WS_ELEMS (OFF_OW2 + (size_t)DY * HID)     // oW2T [32][512]

// ---------------- fast branch-free saturating tanh ----------------
// tanh(x) = 1 - 2/(e^{2x}+1); exp2 path: 1 trans + 1 rcp + 3 valu, no branches.
// Large |x| saturates correctly through inf/0 propagation.
__device__ __forceinline__ float fast_tanh(float x) {
  float t = __builtin_amdgcn_exp2f(x * 2.8853900817779268f);  // e^{2x}
  return 1.0f - 2.0f * __builtin_amdgcn_rcpf(t + 1.0f);
}

// ---------------- fragment helpers ----------------
// CDNA5 16-bit A-matrix 16x32: lanes 0-15 hold K 0..7 (v0-3) and 16..23 (v4-7);
// lanes 16-31 hold K 8..15 and 24..31.  B (32x16) mirrored: lane -> column.
__device__ __forceinline__ v16bf load_frag(const __bf16* p) {
  v8bf lo = *(const v8bf*)(p);
  v8bf hi = *(const v8bf*)(p + 16);
  v16bf r;
#pragma unroll
  for (int e = 0; e < 8; ++e) { r[e] = lo[e]; r[e + 8] = hi[e]; }
  return r;
}

__device__ __forceinline__ v16bf load_a(const __bf16* As, int lda, int kb, int lane) {
  return load_frag(As + (lane & 15) * lda + kb + ((lane < 16) ? 0 : 8));
}

__device__ __forceinline__ v16bf load_b(const __bf16* Wt, int kpad, int n0, int kb, int lane) {
  return load_frag(Wt + (size_t)(n0 + (lane & 15)) * kpad + kb + ((lane < 16) ? 0 : 8));
}

__device__ __forceinline__ v8f wmma_bf16(v16bf a, v16bf b, v8f c) {
  return __builtin_amdgcn_wmma_f32_16x16x32_bf16(false, a, false, b, (short)0, c,
                                                 false, false);
}

// Single-tile accumulate, software pipelined (used for the N=32 output GEMMs)
__device__ __forceinline__ v8f gemm_acc(v8f acc, const __bf16* As, int lda,
                                        const __bf16* Wt, int kpad, int n0,
                                        int kbA, int kbW, int nchunks, int lane) {
  v16bf a = load_a(As, lda, kbA, lane);
  v16bf b = load_b(Wt, kpad, n0, kbW, lane);
#pragma unroll
  for (int kc = 0; kc < nchunks; ++kc) {
    v16bf an = a, bn = b;
    if (kc + 1 < nchunks) {              // prefetch next chunk before WMMA
      an = load_a(As, lda, kbA + (kc + 1) * 32, lane);
      bn = load_b(Wt, kpad, n0, kbW + (kc + 1) * 32, lane);
    }
    acc = wmma_bf16(a, b, acc);
    a = an; b = bn;
  }
  return acc;
}

// Dual-tile accumulate: one A fragment feeds two WMMAs; next chunk's loads are
// issued before the current WMMAs so VMEM/LDS latency hides under matrix ops.
__device__ __forceinline__ void gemm_dual(v8f& acc0, v8f& acc1,
                                          const __bf16* As, int lda,
                                          const __bf16* Wt, int kpad,
                                          int n0, int n1, int kbA, int kbW,
                                          int nchunks, int lane) {
  v16bf a  = load_a(As, lda, kbA, lane);
  v16bf b0 = load_b(Wt, kpad, n0, kbW, lane);
  v16bf b1 = load_b(Wt, kpad, n1, kbW, lane);
#pragma unroll
  for (int kc = 0; kc < nchunks; ++kc) {
    v16bf an = a, b0n = b0, b1n = b1;
    if (kc + 1 < nchunks) {              // prefetch next chunk before WMMAs
      an  = load_a(As, lda, kbA + (kc + 1) * 32, lane);
      b0n = load_b(Wt, kpad, n0, kbW + (kc + 1) * 32, lane);
      b1n = load_b(Wt, kpad, n1, kbW + (kc + 1) * 32, lane);
    }
    acc0 = wmma_bf16(a, b0, acc0);
    acc1 = wmma_bf16(a, b1, acc1);
    a = an; b0 = b0n; b1 = b1n;
  }
}

// bias + tanh + bf16 store epilogue for one 16x16 tile
__device__ __forceinline__ void epi_tanh(__bf16* dst, int ldd, v8f acc,
                                         const float* bias, int n0, int rowbase,
                                         int col) {
  float bb = bias[n0 + col];
#pragma unroll
  for (int j = 0; j < 8; ++j)
    dst[(rowbase + j) * ldd + n0 + col] = (__bf16)fast_tanh(acc[j] + bb);
}

// ---------------- weight convert/transpose kernel ----------------
// src: f32 row-major [K][N] -> dst: bf16 [N][Kpad] (zero padded K..Kpad)
__global__ void njode_convert_w(const float* __restrict__ src,
                                unsigned short* __restrict__ dst_u,
                                int K, int N, int Kpad) {
  int idx = blockIdx.x * blockDim.x + threadIdx.x;
  if (idx >= N * Kpad) return;
  int n = idx / Kpad;
  int k = idx - n * Kpad;
  float v = (k < K) ? src[(size_t)k * N + n] : 0.0f;
  ((__bf16*)dst_u)[idx] = (__bf16)v;
}

// ---------------- main persistent recurrence kernel ----------------
__global__ __launch_bounds__(NTHREADS)
void njode_kernel(const float* __restrict__ times, const float* __restrict__ values,
                  const unsigned short* jW1T_u, const unsigned short* jW2T_u,
                  const unsigned short* fW1T_u, const unsigned short* fW2T_u,
                  const unsigned short* oW1T_u, const unsigned short* oW2T_u,
                  const float* __restrict__ jb1, const float* __restrict__ jb2,
                  const float* __restrict__ fb1, const float* __restrict__ fb2,
                  const float* __restrict__ ob1, const float* __restrict__ ob2,
                  const int* __restrict__ nsteps_p,
                  float* __restrict__ preds, float* __restrict__ preds_before) {
  const __bf16* jW1T = (const __bf16*)jW1T_u;
  const __bf16* jW2T = (const __bf16*)jW2T_u;
  const __bf16* fW1T = (const __bf16*)fW1T_u;
  const __bf16* fW2T = (const __bf16*)fW2T_u;
  const __bf16* oW1T = (const __bf16*)oW1T_u;
  const __bf16* oW2T = (const __bf16*)oW2T_u;

  __shared__ __attribute__((aligned(16))) __bf16 sH[MT * HID];   // hidden state
  __shared__ __attribute__((aligned(16))) __bf16 sA[MT * HID];   // layer-1 activations
  __shared__ __attribute__((aligned(16))) __bf16 sX[MT * DX];    // current observation
  __shared__ float sTi[MT];
  __shared__ float sDt[MT];
  __shared__ float sYprev[MT * DY];                              // output carry

  const int tid  = threadIdx.x;
  const int lane = tid & 31;
  const int wave = tid >> 5;
  const int r0   = blockIdx.x * MT;
  const int nsteps = nsteps_p[0];
  const float inv_nsteps = __builtin_amdgcn_rcpf((float)nsteps);

  const int col     = lane & 15;             // column within 16-wide N tile
  const int rowbase = (lane < 16) ? 0 : 8;   // C/D: vgpr j -> row rowbase+j
  const int n0      = wave * 16;             // dense GEMM tile 0
  const int n1      = n0 + 256;              // dense GEMM tile 1 (nt + 16)

  if (tid < MT * DY) sYprev[tid] = 0.0f;
  __syncthreads();

  for (int i = 0; i < NOBS; ++i) {
    // ---- emit carry (preds_before), load x / t for this step (512 = 16x32) ----
    {
      int row = tid >> 5, c = tid & 31;
      size_t ob = (((size_t)(r0 + row)) * NOBS + i) * DY + c;
      preds_before[ob] = sYprev[row * DY + c];
      sX[row * DX + c] =
          (__bf16)values[(((size_t)(r0 + row)) * NOBS + i) * DX + c];
      if (c == 0) {
        float ti = times[(size_t)(r0 + row) * NOBS + i];
        float tn = times[(size_t)(r0 + row) * NOBS + ((i < NOBS - 1) ? i + 1 : i)];
        sTi[row] = ti;
        sDt[row] = (tn - ti) * inv_nsteps;
      }
    }
    __syncthreads();

    // ---- jump_nn layer 1: tanh(x @ jW1 + jb1) -> sA ----
    {
      v8f a0 = {}, a1 = {};
      gemm_dual(a0, a1, sX, DX, jW1T, DX, n0, n1, 0, 0, DX / 32, lane);
      epi_tanh(sA, HID, a0, jb1, n0, rowbase, col);
      epi_tanh(sA, HID, a1, jb1, n1, rowbase, col);
    }
    __syncthreads();

    // ---- jump_nn layer 2: h = tanh(a1 @ jW2 + jb2) -> sH ----
    {
      v8f a0 = {}, a1 = {};
      gemm_dual(a0, a1, sA, HID, jW2T, HID, n0, n1, 0, 0, HID / 32, lane);
      epi_tanh(sH, HID, a0, jb2, n0, rowbase, col);
      epi_tanh(sH, HID, a1, jb2, n1, rowbase, col);
    }
    __syncthreads();

    // ---- y_i = output_nn(h): o1 = tanh(h @ oW1 + ob1) -> sA ----
    {
      v8f a0 = {}, a1 = {};
      gemm_dual(a0, a1, sH, HID, oW1T, HID, n0, n1, 0, 0, HID / 32, lane);
      epi_tanh(sA, HID, a0, ob1, n0, rowbase, col);
      epi_tanh(sA, HID, a1, ob1, n1, rowbase, col);
    }
    __syncthreads();
    // ---- y_i = o1 @ oW2 + ob2 -> preds (N = 32 -> waves 0,1) ----
    for (int nt = wave; nt < DY / 16; nt += NWAVES) {
      int m0 = nt * 16;
      v8f acc = {};
      acc = gemm_acc(acc, sA, HID, oW2T, HID, m0, 0, 0, HID / 32, lane);
      float bb = ob2[m0 + col];
#pragma unroll
      for (int j = 0; j < 8; ++j)
        preds[(((size_t)(r0 + rowbase + j)) * NOBS + i) * DY + m0 + col] =
            acc[j] + bb;
    }
    __syncthreads();

    // ---- Euler ODE loop: h += dt * (tanh([h|x|t|dt] @ fW1 + fb1) @ fW2 + fb2) ----
    for (int k = 0; k < nsteps; ++k) {
      // layer 1 over K = 546 (split: h 0..511 | x 512..543 | t,dt 544..545, pad)
      {
        v8f a0 = {}, a1 = {};
        gemm_dual(a0, a1, sH, HID, fW1T, KFPAD, n0, n1, 0, 0, HID / 32, lane);
        gemm_dual(a0, a1, sX, DX, fW1T, KFPAD, n0, n1, 0, HID, 1, lane);
        // register-built A chunk for K = 544..575 (t_cur, dt, zeros)
        v16bf at;
#pragma unroll
        for (int e = 0; e < 16; ++e) at[e] = (__bf16)0.0f;
        if (lane < 16) {
          int r = lane & 15;
          at[0] = (__bf16)(sTi[r] + (float)k * sDt[r]);  // K=544 : t_cur
          at[1] = (__bf16)sDt[r];                        // K=545 : t_elapsed
        }
        v16bf bt0 = load_b(fW1T, KFPAD, n0, 544, lane);
        v16bf bt1 = load_b(fW1T, KFPAD, n1, 544, lane);
        a0 = wmma_bf16(at, bt0, a0);
        a1 = wmma_bf16(at, bt1, a1);
        epi_tanh(sA, HID, a0, fb1, n0, rowbase, col);
        epi_tanh(sA, HID, a1, fb1, n1, rowbase, col);
      }
      __syncthreads();
      // layer 2 + fused Euler update of sH
      {
        v8f a0 = {}, a1 = {};
        gemm_dual(a0, a1, sA, HID, fW2T, HID, n0, n1, 0, 0, HID / 32, lane);
        float b0 = fb2[n0 + col];
        float b1 = fb2[n1 + col];
#pragma unroll
        for (int j = 0; j < 8; ++j) {
          int row = rowbase + j;
          float dtr = sDt[row];
          float h0 = (float)sH[row * HID + n0 + col] + dtr * (a0[j] + b0);
          float h1 = (float)sH[row * HID + n1 + col] + dtr * (a1[j] + b1);
          sH[row * HID + n0 + col] = (__bf16)h0;
          sH[row * HID + n1 + col] = (__bf16)h1;
        }
      }
      __syncthreads();
    }

    // ---- y_before_new = output_nn(h_end) -> sYprev (emitted next step) ----
    {
      v8f a0 = {}, a1 = {};
      gemm_dual(a0, a1, sH, HID, oW1T, HID, n0, n1, 0, 0, HID / 32, lane);
      epi_tanh(sA, HID, a0, ob1, n0, rowbase, col);
      epi_tanh(sA, HID, a1, ob1, n1, rowbase, col);
    }
    __syncthreads();
    for (int nt = wave; nt < DY / 16; nt += NWAVES) {
      int m0 = nt * 16;
      v8f acc = {};
      acc = gemm_acc(acc, sA, HID, oW2T, HID, m0, 0, 0, HID / 32, lane);
      float bb = ob2[m0 + col];
#pragma unroll
      for (int j = 0; j < 8; ++j)
        sYprev[(rowbase + j) * DY + m0 + col] = acc[j] + bb;
    }
    __syncthreads();
  }
}

// ---------------- host entry ----------------
extern "C" void kernel_launch(void* const* d_in, const int* in_sizes, int n_in,
                              void* d_out, int out_size, void* d_ws, size_t ws_size,
                              hipStream_t stream) {
  const float* times = (const float*)d_in[0];
  const float* values = (const float*)d_in[1];
  const float* jW1 = (const float*)d_in[2];
  const float* jb1 = (const float*)d_in[3];
  const float* jW2 = (const float*)d_in[4];
  const float* jb2 = (const float*)d_in[5];
  const float* fW1 = (const float*)d_in[6];
  const float* fb1 = (const float*)d_in[7];
  const float* fW2 = (const float*)d_in[8];
  const float* fb2 = (const float*)d_in[9];
  const float* oW1 = (const float*)d_in[10];
  const float* ob1 = (const float*)d_in[11];
  const float* oW2 = (const float*)d_in[12];
  const float* ob2 = (const float*)d_in[13];
  const int* nst = (const int*)d_in[14];

  float* preds = (float*)d_out;
  float* preds_before = preds + (size_t)BATCH * NOBS * DY;
  unsigned short* ws = (unsigned short*)d_ws;

  // pre-convert weights: f32 [K][N] -> bf16 [N][Kpad] (column-major in K)
  struct Cvt { const float* s; size_t off; int K, N, Kpad; };
  const Cvt cv[6] = {
      {jW1, OFF_JW1, DX, HID, DX},      {jW2, OFF_JW2, HID, HID, HID},
      {fW1, OFF_FW1, KF, HID, KFPAD},   {fW2, OFF_FW2, HID, HID, HID},
      {oW1, OFF_OW1, HID, HID, HID},    {oW2, OFF_OW2, HID, DY, HID}};
  for (int w = 0; w < 6; ++w) {
    int total = cv[w].N * cv[w].Kpad;
    njode_convert_w<<<(total + 255) / 256, 256, 0, stream>>>(
        cv[w].s, ws + cv[w].off, cv[w].K, cv[w].N, cv[w].Kpad);
  }

  njode_kernel<<<BATCH / MT, NTHREADS, 0, stream>>>(
      times, values, ws + OFF_JW1, ws + OFF_JW2, ws + OFF_FW1, ws + OFF_FW2,
      ws + OFF_OW1, ws + OFF_OW2, jb1, jb2, fb1, fb2, ob1, ob2, nst, preds,
      preds_before);
}
